// ConflictAttentionModel_6579889898214
// MI455X (gfx1250) — compile-verified, matrix operand order checked by reading, not run on hardware
//
#include <hip/hip_runtime.h>
#include <hip/hip_bf16.h>

// ---------------------------------------------------------------------------
// ConflictAttentionModel fused kernel for MI455X (gfx1250, wave32, WMMA bf16)
// B=512, A=128, N=1024, E=256, H=8, DH=32, HID=512, L=2
// One workgroup (256 threads = 8 waves) per batch element; activations live in
// LDS (bf16), all GEMMs via v_wmma_f32_16x16x32_bf16 with fp32 accumulation.
// ---------------------------------------------------------------------------

typedef __bf16 bf16_t;
typedef __attribute__((ext_vector_type(16))) __bf16 v16bf;
typedef __attribute__((ext_vector_type(8)))  __bf16 v8bf;
typedef __attribute__((ext_vector_type(8)))  float  v8f;
typedef __attribute__((ext_vector_type(4)))  float  v4f;

#define Az   128
#define Nz   1024
#define Ez   256
#define Hh   8
#define DHd  32
#define HIDz 512
#define NEGF (-1.0e9f)

// LDS layout (bytes):
//   X    [128][256] bf16 : 0      .. 65536
//   Ybuf [128][256] bf16 : 65536  .. 131072
//   scr  (union)         : 131072 .. 262144
//   acts [128] int       : 262144 .. 262656
#define SMEM_BYTES 262656

struct LayerP {
  const bf16_t *Wqt, *Wkt, *Wvt, *Wot, *W1t, *W2t;   // transposed bf16 (N-major, K-contig)
  const float  *b1, *b2, *ln1s, *ln1b, *ln2s, *ln2b; // fp32 vectors
};
struct KParams {
  const float* agent;  // [B][128][256] f32
  const float* city;   // [B][1024][256] f32
  const int*   acts;   // [B][128]
  LayerP layer[2];
  const bf16_t *Wqst, *Wkst;
  float* out;          // [B][128][128] f32
};

// --- A/B operand loaders matching the 16-bit 16x32 WMMA VGPR layout ---------
// lanes 0-15: row M=lane,   K = k0+0..7  and k0+16..23
// lanes16-31: row M=lane-16,K = k0+8..15 and k0+24..31
__device__ __forceinline__ v16bf load_a_bf16(const bf16_t* base, int stride,
                                             int m0, int k0, int lane) {
  int mr = m0 + (lane & 15);
  int kk = k0 + ((lane >> 4) << 3);
  const bf16_t* p = base + mr * stride + kk;
  v8bf lo = *(const v8bf*)(p);
  v8bf hi = *(const v8bf*)(p + 16);
  v16bf a;
#pragma unroll
  for (int i = 0; i < 8; ++i) { a[i] = lo[i]; a[i + 8] = hi[i]; }
  return a;
}

__device__ __forceinline__ v16bf load_a_f32(const float* base, int stride,
                                            int m0, int k0, int lane) {
  int mr = m0 + (lane & 15);
  int kk = k0 + ((lane >> 4) << 3);
  const float* p = base + mr * stride + kk;
  v4f a0 = *(const v4f*)(p);
  v4f a1 = *(const v4f*)(p + 4);
  v4f b0 = *(const v4f*)(p + 16);
  v4f b1 = *(const v4f*)(p + 20);
  v16bf a;
#pragma unroll
  for (int i = 0; i < 4; ++i) {
    a[i]      = (bf16_t)a0[i];
    a[i + 4]  = (bf16_t)a1[i];
    a[i + 8]  = (bf16_t)b0[i];
    a[i + 12] = (bf16_t)b1[i];
  }
  return a;
}

__device__ __forceinline__ v8f wmma_bf16(v16bf a, v16bf b, v8f c) {
  return __builtin_amdgcn_wmma_f32_16x16x32_bf16(
      false, a, false, b, (short)0, c, false, false);
}

// C tile (f32 16x16, lane=N%16, VGPR r -> M = r + 8*(lane>=16)) -> bf16 row-major
__device__ __forceinline__ void store_c_bf16(bf16_t* base, int stride,
                                             int m0, int n0, v8f c, int lane) {
  int n  = n0 + (lane & 15);
  int mb = m0 + ((lane >> 4) << 3);
#pragma unroll
  for (int r = 0; r < 8; ++r) base[(mb + r) * stride + n] = (bf16_t)c[r];
}

// transposed store: element (m,n) -> base[n*stride + m]  (for V^T staging)
__device__ __forceinline__ void store_c_bf16_t(bf16_t* base, int stride,
                                               int m0, int n0, v8f c, int lane) {
  int n  = n0 + (lane & 15);
  int mb = m0 + ((lane >> 4) << 3);
#pragma unroll
  for (int r = 0; r < 8; ++r) base[n * stride + mb + r] = (bf16_t)c[r];
}

__device__ __forceinline__ void layer_norm_rows(bf16_t* buf, const float* sc,
                                                const float* bi, int tid) {
  if (tid < Az) {
    bf16_t* p = buf + tid * Ez;
    float s = 0.f, s2 = 0.f;
#pragma unroll 8
    for (int i = 0; i < Ez; ++i) { float v = (float)p[i]; s += v; s2 += v * v; }
    float mu  = s * (1.0f / Ez);
    float var = s2 * (1.0f / Ez) - mu * mu;
    float rs  = rsqrtf(var + 1e-5f);
#pragma unroll 8
    for (int i = 0; i < Ez; ++i) {
      float v = ((float)p[i] - mu) * rs * sc[i] + bi[i];
      p[i] = (bf16_t)v;
    }
  }
}

__global__ __launch_bounds__(256)
void conflict_attn_kernel(KParams P) {
  extern __shared__ char smem[];
  bf16_t* X      = (bf16_t*)(smem);
  bf16_t* Ybuf   = (bf16_t*)(smem + 65536);
  char*   scr    = smem + 131072;
  int*    acts_s = (int*)(smem + 262144);

  const int tid  = threadIdx.x;
  const int wave = tid >> 5;
  const int lane = tid & 31;
  const int b    = blockIdx.x;
  const int q0   = wave * 16;   // this wave's query-row block
  const int nloc = lane & 15;
  const int half = lane >> 4;

  const float* agentB = P.agent + (size_t)b * Az * Ez;

  if (tid < Az) acts_s[tid] = P.acts[(size_t)b * Az + tid];
  __syncthreads();

  // ---- gather cac = city_embed[b, acts, :] into X (bf16) ----
  {
    int row = tid >> 1;
    int c0  = (tid & 1) * 128;
    const float* src = P.city + ((size_t)b * Nz + acts_s[row]) * Ez + c0;
    bf16_t* dst = X + row * Ez + c0;
    const v4f* s4 = (const v4f*)src;
#pragma unroll 4
    for (int i = 0; i < 32; ++i) {
      v4f v = s4[i];
#pragma unroll
      for (int t = 0; t < 4; ++t) dst[i * 4 + t] = (bf16_t)v[t];
    }
  }
  __syncthreads();

  // =========================== transformer layers ===========================
#pragma unroll 1
  for (int l = 0; l < 2; ++l) {
    const LayerP Lp = P.layer[l];
    bf16_t* qh    = (bf16_t*)(scr);          // [128][32]
    bf16_t* kh    = (bf16_t*)(scr + 8192);   // [128][32]
    bf16_t* vt    = (bf16_t*)(scr + 16384);  // [32][128]  (V^T)
    bf16_t* attnb = (bf16_t*)(scr + 24576);  // [128][128] wave-private strips
    bf16_t* obf   = (bf16_t*)(scr + 57344);  // [128][256] concat attn output

#pragma unroll 1
    for (int h = 0; h < Hh; ++h) {
      // ---- Q/K/V head projections: 48 tiles over 8 waves ----
#pragma unroll 1
      for (int t = 0; t < 6; ++t) {
        int idx  = wave * 6 + t;
        int mat  = idx >> 4;          // 0=Q 1=K 2=V
        int tile = idx & 15;
        int rt = tile >> 1, ct = tile & 1;
        const bf16_t* Wt = (mat == 0) ? Lp.Wqt : (mat == 1) ? Lp.Wkt : Lp.Wvt;
        v8f acc = {};
#pragma unroll
        for (int kt = 0; kt < 8; ++kt) {
          v16bf a = (mat == 0) ? load_a_bf16(X, Ez, rt * 16, kt * 32, lane)
                               : load_a_f32(agentB, Ez, rt * 16, kt * 32, lane);
          v16bf bb = load_a_bf16(Wt, Ez, h * 32 + ct * 16, kt * 32, lane);
          acc = wmma_bf16(a, bb, acc);
        }
        if (mat == 0)      store_c_bf16  (qh, DHd, rt * 16, ct * 16, acc, lane);
        else if (mat == 1) store_c_bf16  (kh, DHd, rt * 16, ct * 16, acc, lane);
        else               store_c_bf16_t(vt, Az,  rt * 16, ct * 16, acc, lane);
      }
      __syncthreads();

      // ---- scores (in registers) + masked softmax; wave owns rows q0..q0+15
      v8f s[8];
      {
        v16bf aq = load_a_bf16(qh, DHd, q0, 0, lane);
#pragma unroll
        for (int j = 0; j < 8; ++j) {
          v16bf bk = load_a_bf16(kh, DHd, j * 16, 0, lane);
          v8f z = {};
          s[j] = wmma_bf16(aq, bk, z);
        }
      }
#pragma unroll
      for (int r = 0; r < 8; ++r) {
        int   qrow = q0 + half * 8 + r;
        int   aq   = acts_s[qrow];
        float mx   = -3.0e38f;
#pragma unroll
        for (int j = 0; j < 8; ++j) {
          int  key  = j * 16 + nloc;
          int  ak   = acts_s[key];
          bool keep = (aq == 0) ? (qrow == key) : (aq == ak);
          float v = keep ? s[j][r] * 0.17677669529663687f : NEGF; // 1/sqrt(32)
          s[j][r] = v;
          mx = fmaxf(mx, v);
        }
        mx = fmaxf(mx, __shfl_xor(mx, 1));
        mx = fmaxf(mx, __shfl_xor(mx, 2));
        mx = fmaxf(mx, __shfl_xor(mx, 4));
        mx = fmaxf(mx, __shfl_xor(mx, 8));
        float sum = 0.f;
#pragma unroll
        for (int j = 0; j < 8; ++j) {
          float e = __expf(s[j][r] - mx);
          s[j][r] = e;
          sum += e;
        }
        sum += __shfl_xor(sum, 1);
        sum += __shfl_xor(sum, 2);
        sum += __shfl_xor(sum, 4);
        sum += __shfl_xor(sum, 8);
        float inv = 1.0f / sum;
#pragma unroll
        for (int j = 0; j < 8; ++j) s[j][r] *= inv;
      }
      bf16_t* aw = attnb + q0 * Az;  // wave-private 16x128 strip
#pragma unroll
      for (int j = 0; j < 8; ++j) store_c_bf16(aw, Az, 0, j * 16, s[j], lane);

      // ---- o_head = attn @ V_head  (16x32 per wave) ----
#pragma unroll
      for (int ct = 0; ct < 2; ++ct) {
        v8f o = {};
#pragma unroll
        for (int kt = 0; kt < 4; ++kt) {
          v16bf a  = load_a_bf16(aw, Az, 0, kt * 32, lane);
          v16bf bb = load_a_bf16(vt, Az, ct * 16, kt * 32, lane);
          o = wmma_bf16(a, bb, o);
        }
        store_c_bf16(obf, Ez, q0, h * 32 + ct * 16, o, lane);
      }
      __syncthreads();
    }

    // ---- Wo projection + residual -> Ybuf (pre-LN) ----
#pragma unroll 1
    for (int ct = 0; ct < 16; ++ct) {
      v8f acc = {};
#pragma unroll
      for (int kt = 0; kt < 8; ++kt) {
        v16bf a  = load_a_bf16(obf, Ez, q0, kt * 32, lane);
        v16bf bb = load_a_bf16(Lp.Wot, Ez, ct * 16, kt * 32, lane);
        acc = wmma_bf16(a, bb, acc);
      }
      int n = ct * 16 + nloc;
#pragma unroll
      for (int r = 0; r < 8; ++r) {
        int qrow = q0 + half * 8 + r;
        float yv = acc[r] + (float)X[qrow * Ez + n];
        Ybuf[qrow * Ez + n] = (bf16_t)yv;
      }
    }
    __syncthreads();
    layer_norm_rows(Ybuf, Lp.ln1s, Lp.ln1b, tid);   // h = LN1(x + attn)
    __syncthreads();

    // ---- FFN1: hid = relu(h @ W1 + b1) -> scr [128][512] bf16 ----
    bf16_t* hid = (bf16_t*)scr;
#pragma unroll 1
    for (int ct = 0; ct < 32; ++ct) {
      v8f acc = {};
#pragma unroll
      for (int kt = 0; kt < 8; ++kt) {
        v16bf a  = load_a_bf16(Ybuf, Ez, q0, kt * 32, lane);
        v16bf bb = load_a_bf16(Lp.W1t, Ez, ct * 16, kt * 32, lane);
        acc = wmma_bf16(a, bb, acc);
      }
      int n = ct * 16 + nloc;
      float bias = Lp.b1[n];
#pragma unroll
      for (int r = 0; r < 8; ++r) {
        float v = fmaxf(acc[r] + bias, 0.f);
        hid[(q0 + half * 8 + r) * HIDz + n] = (bf16_t)v;
      }
    }
    __syncthreads();

    // ---- FFN2 + residual -> X (pre-LN) ----
#pragma unroll 1
    for (int ct = 0; ct < 16; ++ct) {
      v8f acc = {};
#pragma unroll
      for (int kt = 0; kt < 16; ++kt) {
        v16bf a  = load_a_bf16(hid, HIDz, q0, kt * 32, lane);
        v16bf bb = load_a_bf16(Lp.W2t, HIDz, ct * 16, kt * 32, lane);
        acc = wmma_bf16(a, bb, acc);
      }
      int n = ct * 16 + nloc;
      float bias = Lp.b2[n];
#pragma unroll
      for (int r = 0; r < 8; ++r) {
        int qrow = q0 + half * 8 + r;
        float v = acc[r] + bias + (float)Ybuf[qrow * Ez + n];
        X[qrow * Ez + n] = (bf16_t)v;
      }
    }
    __syncthreads();
    layer_norm_rows(X, Lp.ln2s, Lp.ln2b, tid);      // x = LN2(h + ff)
    __syncthreads();
  }

  // =============================== final head ===============================
  bf16_t* Qs = (bf16_t*)(scr);            // [128][256]
  bf16_t* Ks = (bf16_t*)(scr + 65536);    // [128][256]
#pragma unroll 1
  for (int ct = 0; ct < 16; ++ct) {
    v8f a1 = {}, a2 = {};
#pragma unroll
    for (int kt = 0; kt < 8; ++kt) {
      v16bf xa = load_a_bf16(X, Ez, q0, kt * 32, lane);
      v16bf wb = load_a_bf16(P.Wqst, Ez, ct * 16, kt * 32, lane);
      a1 = wmma_bf16(xa, wb, a1);
      v16bf ga = load_a_f32(agentB, Ez, q0, kt * 32, lane);
      v16bf kb = load_a_bf16(P.Wkst, Ez, ct * 16, kt * 32, lane);
      a2 = wmma_bf16(ga, kb, a2);
    }
    store_c_bf16(Qs, Ez, q0, ct * 16, a1, lane);
    store_c_bf16(Ks, Ez, q0, ct * 16, a2, lane);
  }
  __syncthreads();

  float* outB = P.out + (size_t)b * Az * Az;
#pragma unroll 1
  for (int j = 0; j < 8; ++j) {
    v8f acc = {};
#pragma unroll
    for (int kt = 0; kt < 8; ++kt) {
      v16bf a  = load_a_bf16(Qs, Ez, q0, kt * 32, lane);
      v16bf bb = load_a_bf16(Ks, Ez, j * 16, kt * 32, lane);
      acc = wmma_bf16(a, bb, acc);
    }
    int key = j * 16 + nloc;
    int ak  = acts_s[key];
#pragma unroll
    for (int r = 0; r < 8; ++r) {
      int  qrow = q0 + half * 8 + r;
      int  aq   = acts_s[qrow];
      bool keep = (aq == 0) ? (qrow == key) : (aq == ak);
      float v = keep ? 10.0f * tanhf(acc[r] * 0.0625f) : NEGF; // 1/sqrt(256)
      outB[qrow * Az + key] = v;
    }
  }
}

// ------------------- weight prep: fp32 [K][N] -> bf16 [N][K] -----------------
__global__ void transpose_to_bf16(const float* __restrict__ in,
                                  bf16_t* __restrict__ out, int K, int N) {
  int i = blockIdx.x * blockDim.x + threadIdx.x;
  if (i >= K * N) return;
  int k = i / N, n = i - k * N;
  out[n * K + k] = (bf16_t)in[i];
}

// ---------------------------------------------------------------------------
extern "C" void kernel_launch(void* const* d_in, const int* in_sizes, int n_in,
                              void* d_out, int out_size, void* d_ws, size_t ws_size,
                              hipStream_t stream) {
  (void)out_size; (void)ws_size;
  // Input flattening: agent_embed, city_embed, acts, then params leaves.
  // Detect jax-pytree (sorted dict keys) vs insertion order via in_sizes[5]:
  //   sorted:    3=Wk_s 4=Wq_s, layer base 5+12l: W1,W2,Wk,Wo,Wq,Wv,b1,b2,ln1_b,ln1_s,ln2_b,ln2_s
  //   insertion: layer base 3+12l: Wq,Wk,Wv,Wo,W1,b1,W2,b2,ln1_s,ln1_b,ln2_s,ln2_b; 27=Wq_s 28=Wk_s
  bool sorted = (n_in > 5 && in_sizes[5] == Ez * HIDz);
  int iWq[2], iWk[2], iWv[2], iWo[2], iW1[2], iW2[2];
  int ib1[2], ib2[2], il1s[2], il1b[2], il2s[2], il2b[2], iWqs, iWks;
  for (int l = 0; l < 2; ++l) {
    if (sorted) {
      int b0 = 5 + 12 * l;
      iW1[l] = b0 + 0;  iW2[l] = b0 + 1;  iWk[l] = b0 + 2;  iWo[l] = b0 + 3;
      iWq[l] = b0 + 4;  iWv[l] = b0 + 5;  ib1[l] = b0 + 6;  ib2[l] = b0 + 7;
      il1b[l] = b0 + 8; il1s[l] = b0 + 9; il2b[l] = b0 + 10; il2s[l] = b0 + 11;
    } else {
      int b0 = 3 + 12 * l;
      iWq[l] = b0 + 0;  iWk[l] = b0 + 1;  iWv[l] = b0 + 2;  iWo[l] = b0 + 3;
      iW1[l] = b0 + 4;  ib1[l] = b0 + 5;  iW2[l] = b0 + 6;  ib2[l] = b0 + 7;
      il1s[l] = b0 + 8; il1b[l] = b0 + 9; il2s[l] = b0 + 10; il2b[l] = b0 + 11;
    }
  }
  if (sorted) { iWks = 3; iWqs = 4; } else { iWqs = 27; iWks = 28; }

  // ws layout (bf16 elements)
  bf16_t* ws = (bf16_t*)d_ws;
  size_t off = 0;
  size_t oWq[2], oWk[2], oWv[2], oWo[2], oW1[2], oW2[2], oWqs, oWks;
  for (int l = 0; l < 2; ++l) {
    oWq[l] = off; off += (size_t)Ez * Ez;
    oWk[l] = off; off += (size_t)Ez * Ez;
    oWv[l] = off; off += (size_t)Ez * Ez;
    oWo[l] = off; off += (size_t)Ez * Ez;
    oW1[l] = off; off += (size_t)Ez * HIDz;
    oW2[l] = off; off += (size_t)HIDz * Ez;
  }
  oWqs = off; off += (size_t)Ez * Ez;
  oWks = off; off += (size_t)Ez * Ez;

  auto tr = [&](int idx, size_t dst, int K, int N) {
    int total = K * N;
    transpose_to_bf16<<<(total + 255) / 256, 256, 0, stream>>>(
        (const float*)d_in[idx], ws + dst, K, N);
  };
  for (int l = 0; l < 2; ++l) {
    tr(iWq[l], oWq[l], Ez, Ez);
    tr(iWk[l], oWk[l], Ez, Ez);
    tr(iWv[l], oWv[l], Ez, Ez);
    tr(iWo[l], oWo[l], Ez, Ez);
    tr(iW1[l], oW1[l], Ez, HIDz);
    tr(iW2[l], oW2[l], HIDz, Ez);
  }
  tr(iWqs, oWqs, Ez, Ez);
  tr(iWks, oWks, Ez, Ez);

  KParams P;
  P.agent = (const float*)d_in[0];
  P.city  = (const float*)d_in[1];
  P.acts  = (const int*)d_in[2];
  for (int l = 0; l < 2; ++l) {
    P.layer[l].Wqt  = ws + oWq[l];
    P.layer[l].Wkt  = ws + oWk[l];
    P.layer[l].Wvt  = ws + oWv[l];
    P.layer[l].Wot  = ws + oWo[l];
    P.layer[l].W1t  = ws + oW1[l];
    P.layer[l].W2t  = ws + oW2[l];
    P.layer[l].b1   = (const float*)d_in[ib1[l]];
    P.layer[l].b2   = (const float*)d_in[ib2[l]];
    P.layer[l].ln1s = (const float*)d_in[il1s[l]];
    P.layer[l].ln1b = (const float*)d_in[il1b[l]];
    P.layer[l].ln2s = (const float*)d_in[il2s[l]];
    P.layer[l].ln2b = (const float*)d_in[il2b[l]];
  }
  P.Wqst = ws + oWqs;
  P.Wkst = ws + oWks;
  P.out  = (float*)d_out;

  (void)hipFuncSetAttribute((const void*)conflict_attn_kernel,
                            hipFuncAttributeMaxDynamicSharedMemorySize,
                            SMEM_BYTES);
  conflict_attn_kernel<<<512, 256, SMEM_BYTES, stream>>>(P);
}
